// SGL_74079595921976
// MI455X (gfx1250) — matrix operand order repaired and checked by reading, not compiled, not run on hardware
//
#include <hip/hip_runtime.h>
#include <math.h>

#define USER_NUM 50000
#define ITEM_NUM 100000
#define NNODES   (USER_NUM + ITEM_NUM)
#define DIM      64
#define INV_TAU  2.0f      // 1 / SSL_TAU
#define SSL_W    0.1f
#define LMBD     0.0001f
#define EPSN     1e-12f
// (1/tau) * log2(e): exp(d/tau) == exp2(d * TAU_LOG2E), one v_mul + one v_exp_f32
#define TAU_LOG2E 2.8853900817779268f

typedef _Float16 v8h  __attribute__((ext_vector_type(8)));
typedef _Float16 v16h __attribute__((ext_vector_type(16)));
typedef _Float16 h2   __attribute__((ext_vector_type(2)));
typedef float    v8f  __attribute__((ext_vector_type(8)));

static __device__ __forceinline__ float wred16(float v) {
  v += __shfl_xor(v, 1); v += __shfl_xor(v, 2);
  v += __shfl_xor(v, 4); v += __shfl_xor(v, 8);
  return v;
}
static __device__ __forceinline__ float wred32(float v) {
  v = wred16(v); v += __shfl_xor(v, 16);
  return v;
}
static __device__ __forceinline__ v16h cat8(v8h lo, v8h hi) {
  return __builtin_shufflevector(lo, hi, 0,1,2,3,4,5,6,7,8,9,10,11,12,13,14,15);
}

// ---------------- propagation kernels ----------------

__global__ void k_init(const float* __restrict__ ue, const float* __restrict__ ie,
                       float* __restrict__ h, float* __restrict__ F) {
  size_t i = (size_t)blockIdx.x * blockDim.x + threadIdx.x;
  size_t total = (size_t)NNODES * DIM;
  if (i >= total) return;
  size_t ub = (size_t)USER_NUM * DIM;
  float v = (i < ub) ? ue[i] : ie[i - ub];
  h[i] = v; F[i] = v;
}

__global__ void k_zero(float* __restrict__ p, size_t n) {
  size_t i = (size_t)blockIdx.x * blockDim.x + threadIdx.x;
  if (i < n) p[i] = 0.0f;
}

// one wave per edge, float2 per lane; scatter via f32 atomics (L2-resident target)
__global__ void k_edges(const float* __restrict__ h, const int* __restrict__ src,
                        const int* __restrict__ dst, const float* __restrict__ w,
                        int E, float* __restrict__ out) {
  long long gid = (long long)blockIdx.x * blockDim.x + threadIdx.x;
  long long e = gid >> 5;
  if (e >= E) return;
  int lane = threadIdx.x & 31;
  int s = src[e], d = dst[e];
  float we = w[e];
  float2 v = ((const float2*)(h + (size_t)s * DIM))[lane];
  float* op = out + (size_t)d * DIM + lane * 2;
  atomicAdd(op,     v.x * we);
  atomicAdd(op + 1, v.y * we);
}

// wave per row: L2-normalize h in place, accumulate into layer-mean buffer F
__global__ void k_norm_acc(float* __restrict__ h, float* __restrict__ F, int n) {
  long long gid = (long long)blockIdx.x * blockDim.x + threadIdx.x;
  int row = (int)(gid >> 5);
  if (row >= n) return;
  int lane = threadIdx.x & 31;
  float2* p = (float2*)(h + (size_t)row * DIM);
  float2 v = p[lane];
  float ss = wred32(v.x * v.x + v.y * v.y);
  float inv = 1.0f / fmaxf(sqrtf(ss), EPSN);
  v.x *= inv; v.y *= inv;
  p[lane] = v;
  float2* f = (float2*)(F + (size_t)row * DIM);
  float2 a = f[lane];
  a.x += v.x; a.y += v.y;
  f[lane] = a;
}

// ---------------- BPR ----------------

__global__ void k_bpr(const float* __restrict__ F, const int* __restrict__ uid,
                      const int* __restrict__ pid, const int* __restrict__ nid,
                      int nS, float scale, float* __restrict__ acc) {
  long long gid = (long long)blockIdx.x * blockDim.x + threadIdx.x;
  int i = (int)(gid >> 5);
  if (i >= nS) return;
  int lane = threadIdx.x & 31;
  float2 u = ((const float2*)(F + (size_t)uid[i] * DIM))[lane];
  float2 p = ((const float2*)(F + ((size_t)USER_NUM + pid[i]) * DIM))[lane];
  float2 q = ((const float2*)(F + ((size_t)USER_NUM + nid[i]) * DIM))[lane];
  u.x *= scale; u.y *= scale;
  p.x *= scale; p.y *= scale;
  q.x *= scale; q.y *= scale;
  float pos = wred32(u.x * p.x + u.y * p.y);
  float neg = wred32(u.x * q.x + u.y * q.y);
  float reg = wred32(u.x * u.x + u.y * u.y + p.x * p.x + p.y * p.y + q.x * q.x + q.y * q.y);
  if (lane == 0) {
    float t = pos - neg;
    float sp = fmaxf(-t, 0.0f) + log1pf(expf(-fabsf(t)));  // -log_sigmoid(t)
    atomicAdd(&acc[0], sp);
    atomicAdd(&acc[1], reg);
  }
}

// ---------------- SSL prep ----------------

// wave per sample: gather row, L2-normalize, emit f32 and/or f16
__global__ void k_gnorm(const float* __restrict__ F, const int* __restrict__ ids,
                        int n, int rowOff, float* __restrict__ o32,
                        _Float16* __restrict__ o16) {
  long long gid = (long long)blockIdx.x * blockDim.x + threadIdx.x;
  int i = (int)(gid >> 5);
  if (i >= n) return;
  int lane = threadIdx.x & 31;
  size_t row = (size_t)rowOff + ids[i];
  float2 v = ((const float2*)(F + row * DIM))[lane];
  float ss = wred32(v.x * v.x + v.y * v.y);
  float inv = 1.0f / fmaxf(sqrtf(ss), EPSN);
  v.x *= inv; v.y *= inv;
  if (o32) ((float2*)(o32 + (size_t)i * DIM))[lane] = v;
  if (o16) { h2 hv = { (_Float16)v.x, (_Float16)v.y };
             ((h2*)(o16 + (size_t)i * DIM))[lane] = hv; }
}

// wave per row: L2-normalize all N rows into f16 key matrix
__global__ void k_allnorm16(const float* __restrict__ F, _Float16* __restrict__ K, int n) {
  long long gid = (long long)blockIdx.x * blockDim.x + threadIdx.x;
  int row = (int)(gid >> 5);
  if (row >= n) return;
  int lane = threadIdx.x & 31;
  float2 v = ((const float2*)(F + (size_t)row * DIM))[lane];
  float ss = wred32(v.x * v.x + v.y * v.y);
  float inv = 1.0f / fmaxf(sqrtf(ss), EPSN);
  h2 hv = { (_Float16)(v.x * inv), (_Float16)(v.y * inv) };
  ((h2*)(K + (size_t)row * DIM))[lane] = hv;
}

__global__ void k_dot(const float* __restrict__ a, const float* __restrict__ b,
                      int n, float* __restrict__ out) {
  long long gid = (long long)blockIdx.x * blockDim.x + threadIdx.x;
  int i = (int)(gid >> 5);
  if (i >= n) return;
  int lane = threadIdx.x & 31;
  float2 x = ((const float2*)(a + (size_t)i * DIM))[lane];
  float2 y = ((const float2*)(b + (size_t)i * DIM))[lane];
  float d = wred32(x.x * y.x + x.y * y.y);
  if (lane == 0) out[i] = d;
}

// ---------------- SSL denominator: fused WMMA GEMM + exp + row-sum ----------------
// Each block owns 16 query rows; 8 waves sweep all 16-column key tiles.
// out[q] = sum_j exp(dot(Q_q, K_j) / tau)

// one 16-column key tile: 4x b128 loads -> 2x WMMA (K=64) -> native exp2 accumulate
static __device__ __forceinline__ void ssl_tile(const _Float16* __restrict__ Kt,
                                                int tile, int m, int hs,
                                                v16h a0, v16h a1,
                                                float accr[8]) {
  const _Float16* krow = Kt + (size_t)(tile * 16 + m) * DIM + hs * 8;
  v16h b0 = cat8(*(const v8h*)(krow),      *(const v8h*)(krow + 16));
  v16h b1 = cat8(*(const v8h*)(krow + 32), *(const v8h*)(krow + 48));
  v8f c = {};
  c = __builtin_amdgcn_wmma_f32_16x16x32_f16(false, a0, false, b0, (short)0, c, false, false);
  c = __builtin_amdgcn_wmma_f32_16x16x32_f16(false, a1, false, b1, (short)0, c, false, false);
#pragma unroll
  for (int r = 0; r < 8; r++)
    accr[r] += __builtin_amdgcn_exp2f(c[r] * TAU_LOG2E);  // |logit/tau| <= ~2.1: native exp ok
}

__global__ void k_sslsum(const _Float16* __restrict__ Q,   // [4096][64], l2-normalized
                         const _Float16* __restrict__ Kt,  // [nrows][64], l2-normalized
                         int nrows, float* __restrict__ out) {
  __shared__ float sums[16];
  int lane  = threadIdx.x & 31;
  int wave  = threadIdx.x >> 5;
  int nwav  = blockDim.x >> 5;
  int qbase = blockIdx.x * 16;
  if (threadIdx.x < 16) sums[threadIdx.x] = 0.0f;
  __syncthreads();

  int m  = lane & 15;       // A: row of the 16x16 tile this lane carries
  int hs = lane >> 4;       // which half-wave (K striping selector)

  // A fragment (16x32 f16, per ISA layout): lanes<16 hold K {0..7,16..23}, lanes>=16 {8..15,24..31}
  const _Float16* qrow = Q + (size_t)(qbase + m) * DIM;
  v16h a0 = cat8(*(const v8h*)(qrow +      hs * 8), *(const v8h*)(qrow + 16 + hs * 8));
  v16h a1 = cat8(*(const v8h*)(qrow + 32 + hs * 8), *(const v8h*)(qrow + 48 + hs * 8));

  float accr[8];
#pragma unroll
  for (int r = 0; r < 8; r++) accr[r] = 0.0f;

  int ntiles = nrows >> 4;
  // 2x unrolled sweep: two independent load->wmma->exp chains in flight per wave
  int t = wave;
  for (; t + nwav < ntiles; t += 2 * nwav) {
    ssl_tile(Kt, t,        m, hs, a0, a1, accr);
    ssl_tile(Kt, t + nwav, m, hs, a0, a1, accr);
  }
  if (t < ntiles)
    ssl_tile(Kt, t, m, hs, a0, a1, accr);

  // rows r+8*hs: sum over the 16 lanes of each half-wave (columns), then LDS across waves
#pragma unroll
  for (int r = 0; r < 8; r++) accr[r] = wred16(accr[r]);
  if ((lane & 15) == 0) {
#pragma unroll
    for (int r = 0; r < 8; r++) atomicAdd(&sums[hs * 8 + r], accr[r]);
  }
  __syncthreads();
  if (threadIdx.x < 16) out[qbase + threadIdx.x] = sums[threadIdx.x];
}

// ---------------- final scalar ----------------

__global__ void k_final(const float* __restrict__ acc,
                        const float* __restrict__ dotu, const float* __restrict__ v2u,
                        const float* __restrict__ doti, const float* __restrict__ v2i,
                        int nS, float* __restrict__ out) {
  __shared__ float red[8];
  float s = 0.0f;
  for (int i = threadIdx.x; i < nS; i += blockDim.x) {
    s += logf(v2u[i]) - dotu[i] * INV_TAU;   // -log(v1/v2) per user sample
    s += logf(v2i[i]) - doti[i] * INV_TAU;   // per item sample
  }
  s = wred32(s);
  int lane = threadIdx.x & 31, wave = threadIdx.x >> 5;
  if (lane == 0) red[wave] = s;
  __syncthreads();
  if (threadIdx.x == 0) {
    float ssl = 0.0f;
    for (int wv = 0; wv < (int)(blockDim.x >> 5); wv++) ssl += red[wv];
    float bpr = acc[0] / nS + LMBD * (acc[1] * 0.5f) / nS;
    out[0] = bpr + SSL_W * ssl;
  }
}

// ---------------- host ----------------

extern "C" void kernel_launch(void* const* d_in, const int* in_sizes, int n_in,
                              void* d_out, int out_size, void* d_ws, size_t ws_size,
                              hipStream_t stream) {
  const float* user_emb = (const float*)d_in[0];
  const float* item_emb = (const float*)d_in[1];
  const int*   src  = (const int*)d_in[2];
  const int*   dst  = (const int*)d_in[3];
  const float* w    = (const float*)d_in[4];
  const int*   src1 = (const int*)d_in[5];
  const int*   dst1 = (const int*)d_in[6];
  const float* w1   = (const float*)d_in[7];
  const int*   src2 = (const int*)d_in[8];
  const int*   dst2 = (const int*)d_in[9];
  const float* w2   = (const float*)d_in[10];
  const int*   user_id = (const int*)d_in[11];
  const int*   item_id = (const int*)d_in[12];
  const int*   neg_id  = (const int*)d_in[13];
  int E0 = in_sizes[2], E1 = in_sizes[5], E2 = in_sizes[8], nS = in_sizes[11];

  size_t ND = (size_t)NNODES * DIM;
  char* base = (char*)d_ws;
  size_t off = 0;
  auto alloc = [&](size_t bytes) -> void* {
    off = (off + 255) & ~(size_t)255;
    void* p = base + off; off += bytes; return p;
  };
  float*     hA   = (float*)alloc(ND * 4);
  float*     hB   = (float*)alloc(ND * 4);
  float*     F    = (float*)alloc(ND * 4);
  _Float16*  K16  = (_Float16*)alloc(ND * 2);
  _Float16*  qu16 = (_Float16*)alloc((size_t)nS * DIM * 2);
  _Float16*  qi16 = (_Float16*)alloc((size_t)nS * DIM * 2);
  float*     qu32 = (float*)alloc((size_t)nS * DIM * 4);
  float*     qi32 = (float*)alloc((size_t)nS * DIM * 4);
  float*     ue2  = (float*)alloc((size_t)nS * DIM * 4);
  float*     ie2  = (float*)alloc((size_t)nS * DIM * 4);
  float*     dotu = (float*)alloc((size_t)nS * 4);
  float*     doti = (float*)alloc((size_t)nS * 4);
  float*     v2u  = (float*)alloc((size_t)nS * 4);
  float*     v2i  = (float*)alloc((size_t)nS * 4);
  float*     acc  = (float*)alloc(2 * 4);

  const int T = 256;
  auto nb = [&](size_t t) { return (unsigned)((t + T - 1) / T); };

  auto propagate = [&](const int* s, const int* d, const float* ew, int E, int L) {
    k_init<<<nb(ND), T, 0, stream>>>(user_emb, item_emb, hA, F);
    float* cur = hA; float* nxt = hB;
    for (int l = 0; l < L; l++) {
      k_zero<<<nb(ND), T, 0, stream>>>(nxt, ND);
      k_edges<<<nb((size_t)E * 32), T, 0, stream>>>(cur, s, d, ew, E, nxt);
      k_norm_acc<<<nb((size_t)NNODES * 32), T, 0, stream>>>(nxt, F, NNODES);
      float* tmp = cur; cur = nxt; nxt = tmp;
    }
  };

  k_zero<<<1, T, 0, stream>>>(acc, 2);

  // main graph: MAIN_LAYERS = 4, layer mean = F/5 (applied as scale in BPR gather)
  propagate(src, dst, w, E0, 4);
  k_bpr<<<nb((size_t)nS * 32), T, 0, stream>>>(F, user_id, item_id, neg_id, nS, 0.2f, acc);

  // augmented view 1 (3 layers); l2norm makes the /4 layer-mean scale irrelevant
  propagate(src1, dst1, w1, E1, 3);
  k_gnorm<<<nb((size_t)nS * 32), T, 0, stream>>>(F, user_id, nS, 0,        qu32, qu16);
  k_gnorm<<<nb((size_t)nS * 32), T, 0, stream>>>(F, item_id, nS, USER_NUM, qi32, qi16);

  // augmented view 2 (3 layers)
  propagate(src2, dst2, w2, E2, 3);
  k_gnorm<<<nb((size_t)nS * 32), T, 0, stream>>>(F, user_id, nS, 0,        ue2, (_Float16*)nullptr);
  k_gnorm<<<nb((size_t)nS * 32), T, 0, stream>>>(F, item_id, nS, USER_NUM, ie2, (_Float16*)nullptr);
  k_allnorm16<<<nb((size_t)NNODES * 32), T, 0, stream>>>(F, K16, NNODES);

  // positive-pair logits (f32)
  k_dot<<<nb((size_t)nS * 32), T, 0, stream>>>(qu32, ue2, nS, dotu);
  k_dot<<<nb((size_t)nS * 32), T, 0, stream>>>(qi32, ie2, nS, doti);

  // InfoNCE denominators: WMMA GEMM + exp + row-sum (keys L2-resident)
  k_sslsum<<<nS / 16, 256, 0, stream>>>(qu16, K16, USER_NUM, v2u);
  k_sslsum<<<nS / 16, 256, 0, stream>>>(qi16, K16 + (size_t)USER_NUM * DIM, ITEM_NUM, v2i);

  k_final<<<1, 256, 0, stream>>>(acc, dotu, v2u, doti, v2i, nS, (float*)d_out);
}
